// X2HAttention_66864050864786
// MI455X (gfx1250) — matrix operand order repaired and checked by reading, not compiled, not run on hardware
//
#include <hip/hip_runtime.h>
#include <hip/hip_bf16.h>

#define NN 50000
#define NE 800000

typedef __bf16 v16bf __attribute__((ext_vector_type(16)));
typedef float  v8f   __attribute__((ext_vector_type(8)));

__device__ __forceinline__ v8f wmma_bf16(v16bf a, v16bf b, v8f c) {
  return __builtin_amdgcn_wmma_f32_16x16x32_bf16(false, a, false, b, (short)0, c,
                                                 false, false);
}

// order-preserving float -> uint key (for atomicMax-based segment max)
__device__ __forceinline__ unsigned fkey(float f) {
  unsigned u = __float_as_uint(f);
  return u ^ ((unsigned)(((int)u) >> 31) | 0x80000000u);
}
__device__ __forceinline__ float fdecode(unsigned u) {
  unsigned m = ((int)u >= 0) ? 0xFFFFFFFFu : 0x80000000u;
  return __uint_as_float(u ^ m);
}

// Store one element of a 16x32-per-kstep bf16 A fragment (WMMA A layout).
// frag layout: [(kt*32 + lane)*16 + j], lane = row + 16*half_a
__device__ __forceinline__ void store_a_elem(__bf16* frag, int r, int k, float v) {
  int kt = k >> 5, kk = k & 31;
  int grp = kk >> 3;
  int ha = grp & 1;
  int jj = (kk & 7) + 8 * (grp >> 1);
  frag[(kt * 32 + r + 16 * ha) * 16 + jj] = (__bf16)v;
}

// wave-level 2-layer MLP on a 16-row tile:
//   acc = relu(layernorm(A @ W1 + b1) * g + bt) @ W2 + b2
// A1: KT1 k-steps of A fragments in LDS. A2: 2048 bf16 LDS scratch (hidden frags).
__device__ __forceinline__ void wave_mlp(
    const __bf16* A1, __bf16* A2,
    const __bf16* __restrict__ pw1, const float* __restrict__ b1,
    const float* __restrict__ gw, const float* __restrict__ bw,
    const __bf16* __restrict__ pw2, const float* __restrict__ b2,
    int KT1, v8f acc[8]) {
  const int lane = threadIdx.x & 31;
  const int c = lane & 15;
  const int hf = lane >> 4;
  v8f vz = {};
  v8f hc[8];
#pragma unroll
  for (int nt = 0; nt < 8; ++nt) hc[nt] = vz;
  for (int kt = 0; kt < KT1; ++kt) {
    v16bf a = *(const v16bf*)(A1 + (kt * 32 + lane) * 16);
    const __bf16* bp = pw1 + (size_t)kt * 4096 + lane * 16;
#pragma unroll
    for (int nt = 0; nt < 8; ++nt) {
      v16bf b = *(const v16bf*)(bp + nt * 512);
      hc[nt] = wmma_bf16(a, b, hc[nt]);
    }
  }
#pragma unroll
  for (int nt = 0; nt < 8; ++nt) {
    float bb = b1[nt * 16 + c];
#pragma unroll
    for (int i = 0; i < 8; ++i) hc[nt][i] += bb;
  }
  // LayerNorm over 128 per row (row m = i + 8*hf, cols across 16 lanes x 8 nt)
  float mean[8], rstd[8];
#pragma unroll
  for (int i = 0; i < 8; ++i) {
    float s = 0.f;
#pragma unroll
    for (int nt = 0; nt < 8; ++nt) s += hc[nt][i];
    s += __shfl_xor(s, 1, 32); s += __shfl_xor(s, 2, 32);
    s += __shfl_xor(s, 4, 32); s += __shfl_xor(s, 8, 32);
    mean[i] = s * (1.0f / 128.0f);
  }
#pragma unroll
  for (int i = 0; i < 8; ++i) {
    float s = 0.f;
#pragma unroll
    for (int nt = 0; nt < 8; ++nt) { float t = hc[nt][i] - mean[i]; s += t * t; }
    s += __shfl_xor(s, 1, 32); s += __shfl_xor(s, 2, 32);
    s += __shfl_xor(s, 4, 32); s += __shfl_xor(s, 8, 32);
    rstd[i] = rsqrtf(s * (1.0f / 128.0f) + 1e-5f);
  }
  // normalize + relu, write hidden as A fragments for GEMM2
#pragma unroll
  for (int nt = 0; nt < 8; ++nt) {
    float g = gw[nt * 16 + c], bt = bw[nt * 16 + c];
    int kk = (nt & 1) * 16 + c;
    int grp = kk >> 3;
    int ha = grp & 1;
    int jj = (kk & 7) + 8 * (grp >> 1);
    int kt2 = nt >> 1;
#pragma unroll
    for (int i = 0; i < 8; ++i) {
      int m = i + 8 * hf;
      float y = (hc[nt][i] - mean[i]) * rstd[i] * g + bt;
      y = fmaxf(y, 0.f);
      A2[(kt2 * 32 + m + 16 * ha) * 16 + jj] = (__bf16)y;
    }
  }
  __syncthreads();
#pragma unroll
  for (int nt = 0; nt < 8; ++nt) acc[nt] = vz;
#pragma unroll
  for (int kt = 0; kt < 4; ++kt) {
    v16bf a = *(const v16bf*)(A2 + (kt * 32 + lane) * 16);
    const __bf16* bp = pw2 + (size_t)kt * 4096 + lane * 16;
#pragma unroll
    for (int nt = 0; nt < 8; ++nt) {
      v16bf b = *(const v16bf*)(bp + nt * 512);
      acc[nt] = wmma_bf16(a, b, acc[nt]);
    }
  }
#pragma unroll
  for (int nt = 0; nt < 8; ++nt) {
    float bb = b2[nt * 16 + c];
#pragma unroll
    for (int i = 0; i < 8; ++i) acc[nt][i] += bb;
  }
}

// ---- weight packing: f32 [K,128] row-major -> bf16 WMMA B fragments ----
__global__ void pack_w_kernel(const float* __restrict__ src, __bf16* __restrict__ dst,
                              int K, int KT) {
  int idx = blockIdx.x * 256 + threadIdx.x;
  int total = KT * 8 * 32 * 16;
  if (idx >= total) return;
  int j = idx & 15;
  int lane = (idx >> 4) & 31;
  int nt = (idx >> 9) & 7;
  int kt = idx >> 12;
  int k = kt * 32 + ((j >> 3) << 4) + ((lane >> 4) << 3) + (j & 7);
  int n = nt * 16 + (lane & 15);
  float v = (k < K) ? src[(size_t)k * 128 + n] : 0.f;
  dst[idx] = (__bf16)v;
}

__global__ void init_kernel(unsigned* __restrict__ smax, float* __restrict__ ssum,
                            float* __restrict__ accum) {
  int i = blockIdx.x * 256 + threadIdx.x;
  if (i < NN * 16) { smax[i] = 0u; ssum[i] = 0.f; }
  if (i < NN * 128) accum[i] = 0.f;
}

// ---- per-node MLP (q) : h[N,128] -> q[N,128], 4 waves / block ----
__global__ __launch_bounds__(128) void qnode_kernel(
    const float* __restrict__ hh,
    const __bf16* __restrict__ pw1, const float* __restrict__ b1,
    const float* __restrict__ g, const float* __restrict__ bt,
    const __bf16* __restrict__ pw2, const float* __restrict__ b2,
    float* __restrict__ qout) {
  __shared__ alignas(64) __bf16 A1[4][4 * 512];
  __shared__ alignas(64) __bf16 A2[4][2048];
  const int wave = threadIdx.x >> 5;
  const int lane = threadIdx.x & 31;
  const int c = lane & 15;
  const int hf = lane >> 4;
  int tile = blockIdx.x * 4 + wave;
  if (tile > 3124) tile = 3124;
  const int nb = tile * 16;
  __bf16* frag = A1[wave];
  const float* hr = hh + (size_t)(nb + c) * 128;
  for (int t = hf * 64; t < hf * 64 + 64; ++t) store_a_elem(frag, c, t, hr[t]);
  __syncthreads();
  v8f acc[8];
  wave_mlp(frag, A2[wave], pw1, b1, g, bt, pw2, b2, 4, acc);
#pragma unroll
  for (int nt = 0; nt < 8; ++nt)
#pragma unroll
    for (int i = 0; i < 8; ++i) {
      int m = i + 8 * hf;
      qout[(size_t)(nb + m) * 128 + nt * 16 + c] = acc[nt][i];
    }
}

// ---- per-edge: build KV feats, hk/hv MLPs, scores + segment-max, v store ----
__global__ __launch_bounds__(128) void edge_kernel(
    const float* __restrict__ xx, const float* __restrict__ hh,
    const float* __restrict__ ea, const int* __restrict__ ei,
    const __bf16* __restrict__ pk1, const float* __restrict__ kb1,
    const float* __restrict__ kg, const float* __restrict__ kbt,
    const __bf16* __restrict__ pk2, const float* __restrict__ kb2,
    const __bf16* __restrict__ pv1, const float* __restrict__ vb1,
    const float* __restrict__ vg, const float* __restrict__ vbt,
    const __bf16* __restrict__ pv2, const float* __restrict__ vb2,
    const float* __restrict__ eww, const float* __restrict__ ewb,
    const float* __restrict__ q, float* __restrict__ scores,
    unsigned* __restrict__ smax, __bf16* __restrict__ vbuf) {
  __shared__ alignas(64) __bf16 A1[4][11 * 512];
  __shared__ alignas(64) __bf16 A2[4][2048];   // hidden frags / k-tile dump
  __shared__ float EW[4][16];

  const int wave = threadIdx.x >> 5;
  const int lane = threadIdx.x & 31;
  const int c = lane & 15;
  const int hf = lane >> 4;
  const int eb = (blockIdx.x * 4 + wave) * 16;
  const int e = eb + c;
  const int s0 = ei[e], d0 = ei[NE + e];

  float dx = xx[d0 * 3 + 0] - xx[s0 * 3 + 0];
  float dy = xx[d0 * 3 + 1] - xx[s0 * 3 + 1];
  float dz = xx[d0 * 3 + 2] - xx[s0 * 3 + 2];
  float dist = sqrtf(dx * dx + dy * dy + dz * dz);
  const float step = 10.0f / 19.0f;
  const float coeff = -0.5f / (step * step);
  float gs[20];
#pragma unroll
  for (int gi = 0; gi < 20; ++gi) {
    float d = dist - step * (float)gi;
    gs[gi] = __expf(coeff * d * d);
  }
  float ea0 = ea[e * 4 + 0], ea1 = ea[e * 4 + 1];
  float ea2 = ea[e * 4 + 2], ea3 = ea[e * 4 + 3];
  __bf16* frag = A1[wave];
  if (hf == 0) {                 // features 0..175 + edge weight
    store_a_elem(frag, c, 0, ea0);
    store_a_elem(frag, c, 1, ea1);
    store_a_elem(frag, c, 2, ea2);
    store_a_elem(frag, c, 3, ea3);
#pragma unroll
    for (int ef = 0; ef < 4; ++ef) {
      float a = ef == 0 ? ea0 : ef == 1 ? ea1 : ef == 2 ? ea2 : ea3;
#pragma unroll
      for (int gi = 0; gi < 20; ++gi)
        store_a_elem(frag, c, 4 + ef * 20 + gi, a * gs[gi]);
    }
    const float* hi = hh + (size_t)d0 * 128;
    for (int t = 0; t < 92; ++t) store_a_elem(frag, c, 84 + t, hi[t]);
    float s = ewb[0];
#pragma unroll
    for (int ef = 0; ef < 4; ++ef) {
      float a = ef == 0 ? ea0 : ef == 1 ? ea1 : ef == 2 ? ea2 : ea3;
#pragma unroll
      for (int gi = 0; gi < 20; ++gi) s += a * gs[gi] * eww[ef * 20 + gi];
    }
    EW[wave][c] = 1.0f / (1.0f + __expf(-s));
  } else {                       // features 176..351
    const float* hi = hh + (size_t)d0 * 128;
    for (int t = 92; t < 128; ++t) store_a_elem(frag, c, 84 + t, hi[t]);
    const float* hj = hh + (size_t)s0 * 128;
    for (int t = 0; t < 128; ++t) store_a_elem(frag, c, 212 + t, hj[t]);
#pragma unroll
    for (int k = 340; k < 352; ++k) store_a_elem(frag, c, k, 0.f);
  }
  __syncthreads();

  // ---- k = hk MLP ----
  v8f kacc[8];
  wave_mlp(frag, A2[wave], pk1, kb1, kg, kbt, pk2, kb2, 11, kacc);
  __bf16* ksb = A2[wave];        // reuse as [16][128] bf16 k tile
#pragma unroll
  for (int nt = 0; nt < 8; ++nt)
#pragma unroll
    for (int i = 0; i < 8; ++i)
      ksb[(i + 8 * hf) * 128 + nt * 16 + c] = (__bf16)kacc[nt][i];
  __syncthreads();

  // ---- scores + segment-max (lanes 0..15: one edge each) ----
  if (hf == 0) {
    const float* qrow = q + (size_t)d0 * 128;
#pragma unroll
    for (int hd = 0; hd < 16; ++hd) {
      float s = 0.f;
#pragma unroll
      for (int d = 0; d < 8; ++d)
        s += qrow[hd * 8 + d] * (float)ksb[c * 128 + hd * 8 + d];
      s *= 0.3535533905932738f;  // 1/sqrt(8)
      scores[(size_t)e * 16 + hd] = s;
      atomicMax(&smax[(size_t)d0 * 16 + hd], fkey(s));
    }
  }
  __syncthreads();

  // ---- v = hv MLP, scaled by edge weight, stored bf16 ----
  v8f vacc[8];
  wave_mlp(frag, A2[wave], pv1, vb1, vg, vbt, pv2, vb2, 11, vacc);
#pragma unroll
  for (int nt = 0; nt < 8; ++nt)
#pragma unroll
    for (int i = 0; i < 8; ++i) {
      int m = i + 8 * hf;
      float vv = vacc[nt][i] * EW[wave][m];
      vbuf[(size_t)(eb + m) * 128 + nt * 16 + c] = (__bf16)vv;
    }
}

// ---- exp / segment-sum / weighted scatter aggregation ----
__global__ __launch_bounds__(256) void scatter_kernel(
    const int* __restrict__ ei, const float* __restrict__ scores,
    const unsigned* __restrict__ smax, const __bf16* __restrict__ vbuf,
    float* __restrict__ ssum, float* __restrict__ accum) {
  int idx = blockIdx.x * 256 + threadIdx.x;
  int e = idx >> 2, part = idx & 3;
  if (e >= NE) return;
  int dst = ei[NE + e];
  float ex[4];
#pragma unroll
  for (int t = 0; t < 4; ++t) {
    int hd = part * 4 + t;
    float s = scores[(size_t)e * 16 + hd];
    float mx = fdecode(smax[(size_t)dst * 16 + hd]);
    float v = __expf(s - mx);
    ex[t] = v;
    atomicAdd(&ssum[(size_t)dst * 16 + hd], v);
  }
#pragma unroll
  for (int t = 0; t < 4; ++t)
#pragma unroll
    for (int d = 0; d < 8; ++d) {
      int col = part * 32 + t * 8 + d;
      float vv = (float)vbuf[(size_t)e * 128 + col];
      atomicAdd(&accum[(size_t)dst * 128 + col], ex[t] * vv);
    }
}

// ---- node output MLP on [agg, h] (256 feats) + residual ----
__global__ __launch_bounds__(128) void out_kernel(
    const float* __restrict__ hh, const float* __restrict__ accum,
    const float* __restrict__ ssum,
    const __bf16* __restrict__ pw1, const float* __restrict__ b1,
    const float* __restrict__ g, const float* __restrict__ bt,
    const __bf16* __restrict__ pw2, const float* __restrict__ b2,
    float* __restrict__ out) {
  __shared__ alignas(64) __bf16 A1[4][8 * 512];
  __shared__ alignas(64) __bf16 A2[4][2048];
  const int wave = threadIdx.x >> 5;
  const int lane = threadIdx.x & 31;
  const int c = lane & 15;
  const int hf = lane >> 4;
  int tile = blockIdx.x * 4 + wave;
  if (tile > 3124) tile = 3124;
  const int nb = tile * 16;
  const int n = nb + c;
  __bf16* frag = A1[wave];
  for (int t = hf * 128; t < hf * 128 + 128; ++t) {
    float v;
    if (t < 128) {
      float den = ssum[(size_t)n * 16 + (t >> 3)];
      v = (den > 0.f) ? accum[(size_t)n * 128 + t] / den : 0.f;
    } else {
      v = hh[(size_t)n * 128 + (t - 128)];
    }
    store_a_elem(frag, c, t, v);
  }
  __syncthreads();
  v8f acc[8];
  wave_mlp(frag, A2[wave], pw1, b1, g, bt, pw2, b2, 8, acc);
#pragma unroll
  for (int nt = 0; nt < 8; ++nt)
#pragma unroll
    for (int i = 0; i < 8; ++i) {
      int m = i + 8 * hf;
      size_t o = (size_t)(nb + m) * 128 + nt * 16 + c;
      out[o] = acc[nt][i] + hh[o];
    }
}

extern "C" void kernel_launch(void* const* d_in, const int* in_sizes, int n_in,
                              void* d_out, int out_size, void* d_ws, size_t ws_size,
                              hipStream_t stream) {
  (void)in_sizes; (void)n_in; (void)out_size; (void)ws_size;
  const float* x     = (const float*)d_in[0];
  const float* h     = (const float*)d_in[1];
  const float* ea    = (const float*)d_in[2];
  const int*   ei    = (const int*)d_in[3];
  const float* hk_w1 = (const float*)d_in[4];
  const float* hk_b1 = (const float*)d_in[5];
  const float* hk_g  = (const float*)d_in[6];
  const float* hk_bt = (const float*)d_in[7];
  const float* hk_w2 = (const float*)d_in[8];
  const float* hk_b2 = (const float*)d_in[9];
  const float* hv_w1 = (const float*)d_in[10];
  const float* hv_b1 = (const float*)d_in[11];
  const float* hv_g  = (const float*)d_in[12];
  const float* hv_bt = (const float*)d_in[13];
  const float* hv_w2 = (const float*)d_in[14];
  const float* hv_b2 = (const float*)d_in[15];
  const float* hq_w1 = (const float*)d_in[16];
  const float* hq_b1 = (const float*)d_in[17];
  const float* hq_g  = (const float*)d_in[18];
  const float* hq_bt = (const float*)d_in[19];
  const float* hq_w2 = (const float*)d_in[20];
  const float* hq_b2 = (const float*)d_in[21];
  const float* ew_w  = (const float*)d_in[22];
  const float* ew_b  = (const float*)d_in[23];
  const float* no_w1 = (const float*)d_in[24];
  const float* no_b1 = (const float*)d_in[25];
  const float* no_g  = (const float*)d_in[26];
  const float* no_bt = (const float*)d_in[27];
  const float* no_w2 = (const float*)d_in[28];
  const float* no_b2 = (const float*)d_in[29];

  char* ws = (char*)d_ws;
  size_t off = 0;
  auto carve = [&](size_t bytes) -> void* {
    void* p = ws + off;
    off += (bytes + 255) & ~(size_t)255;
    return p;
  };
  __bf16* pk_hk1 = (__bf16*)carve((size_t)11 * 8 * 512 * 2);
  __bf16* pk_hv1 = (__bf16*)carve((size_t)11 * 8 * 512 * 2);
  __bf16* pk_hk2 = (__bf16*)carve((size_t)4 * 8 * 512 * 2);
  __bf16* pk_hv2 = (__bf16*)carve((size_t)4 * 8 * 512 * 2);
  __bf16* pk_hq1 = (__bf16*)carve((size_t)4 * 8 * 512 * 2);
  __bf16* pk_hq2 = (__bf16*)carve((size_t)4 * 8 * 512 * 2);
  __bf16* pk_no1 = (__bf16*)carve((size_t)8 * 8 * 512 * 2);
  __bf16* pk_no2 = (__bf16*)carve((size_t)4 * 8 * 512 * 2);
  float*    qbuf   = (float*)carve((size_t)NN * 128 * 4);
  float*    scores = (float*)carve((size_t)NE * 16 * 4);
  __bf16*   vbuf   = (__bf16*)carve((size_t)NE * 128 * 2);
  unsigned* smax   = (unsigned*)carve((size_t)NN * 16 * 4);
  float*    ssum   = (float*)carve((size_t)NN * 16 * 4);
  float*    accum  = (float*)carve((size_t)NN * 128 * 4);

  pack_w_kernel<<<176, 256, 0, stream>>>(hk_w1, pk_hk1, 340, 11);
  pack_w_kernel<<<176, 256, 0, stream>>>(hv_w1, pk_hv1, 340, 11);
  pack_w_kernel<<<64, 256, 0, stream>>>(hk_w2, pk_hk2, 128, 4);
  pack_w_kernel<<<64, 256, 0, stream>>>(hv_w2, pk_hv2, 128, 4);
  pack_w_kernel<<<64, 256, 0, stream>>>(hq_w1, pk_hq1, 128, 4);
  pack_w_kernel<<<64, 256, 0, stream>>>(hq_w2, pk_hq2, 128, 4);
  pack_w_kernel<<<128, 256, 0, stream>>>(no_w1, pk_no1, 256, 8);
  pack_w_kernel<<<64, 256, 0, stream>>>(no_w2, pk_no2, 128, 4);
  init_kernel<<<25000, 256, 0, stream>>>(smax, ssum, accum);

  qnode_kernel<<<782, 128, 0, stream>>>(h, pk_hq1, hq_b1, hq_g, hq_bt, pk_hq2,
                                        hq_b2, qbuf);
  edge_kernel<<<12500, 128, 0, stream>>>(x, h, ea, ei,
                                         pk_hk1, hk_b1, hk_g, hk_bt, pk_hk2, hk_b2,
                                         pk_hv1, hv_b1, hv_g, hv_bt, pk_hv2, hv_b2,
                                         ew_w, ew_b, qbuf, scores, smax, vbuf);
  scatter_kernel<<<12500, 256, 0, stream>>>(ei, scores, smax, vbuf, ssum, accum);
  out_kernel<<<782, 128, 0, stream>>>(h, accum, ssum, pk_no1, no_b1, no_g, no_bt,
                                      pk_no2, no_b2, (float*)d_out);
}
